// ExternalKnowledge_42477226557526
// MI455X (gfx1250) — matrix operand order repaired and checked by reading, not compiled, not run on hardware
//
#include <hip/hip_runtime.h>

// Problem constants (from reference)
#define B_  32
#define M_  2048
#define T_  4
#define E_  256
#define V_  32000
#define CM_ 128          // CONV_MAX
#define NT_ 16           // M_/128 row-tiles of 128 per batch

typedef __attribute__((ext_vector_type(16))) _Float16 v16h;
typedef __attribute__((ext_vector_type(8)))  float    v8f;

__device__ __forceinline__ float4 ld4(const float* p) {
    return *reinterpret_cast<const float4*>(p);
}
__device__ __forceinline__ void acc4(float4& a, float4 b) {
    a.x += b.x; a.y += b.y; a.z += b.z; a.w += b.w;
}

// ---------------------------------------------------------------------------
// dot_kernel: prob_origin[b,m] = sum_e (embsum_h[b,m,e]+conv) * gp[b,m] * q[b,e]
// One block = (128 rows of one batch). 8 waves; each wave owns a 16-row tile
// and accumulates the dot via v_wmma_f32_16x16x32_f16 (query broadcast into
// all 16 N-columns; f32 accumulate). Gathers are 128-bit, fragment built
// directly in registers following the ISA 16-bit A 16x32 layout:
//   lane L: row r=L%16, half h=L/16; per 32-K chunk needs K in
//   [8h,8h+8) U [16+8h,16+8h+8)  -> two contiguous float4-pair runs.
// ---------------------------------------------------------------------------
__global__ __launch_bounds__(256)
void dot_kernel(const float* __restrict__ Ch,       // C + h*V*E
                const int*   __restrict__ story,    // [B,M,T]
                const int*   __restrict__ kb_len,   // [B]
                const int*   __restrict__ conv_len, // [B]
                const float* __restrict__ rnn_out,  // [B,CM,E]
                const float* __restrict__ gp,       // [B,M]
                const float* __restrict__ q,        // [B,E]
                float*       __restrict__ prob_origin, // [B,M] (ws)
                float*       __restrict__ out_origin)  // d_out half, or null
{
    const int b    = blockIdx.y;
    const int wave = threadIdx.x >> 5;
    const int lane = threadIdx.x & 31;
    const int half = lane >> 4;
    const int r    = lane & 15;
    const int m    = blockIdx.x * 128 + wave * 16 + r;

    const int4  tv   = *reinterpret_cast<const int4*>(story + ((size_t)b * M_ + m) * T_);
    const int   tk[4] = { tv.x, tv.y, tv.z, tv.w };
    const float gpv  = gp[b * M_ + m];
    const int   cidx = m - kb_len[b];
    const bool  inconv = (cidx >= 0) && (cidx < conv_len[b]);
    const float* rrow = rnn_out + ((size_t)b * CM_ + (inconv ? cidx : 0)) * E_;
    const float* qb   = q + b * E_;

    v8f acc = {};
#pragma unroll
    for (int c = 0; c < 8; ++c) {
        const int kb = c * 32;
        const int e0 = kb + 8 * half;        // first K-run of this lane
        const int e1 = kb + 16 + 8 * half;   // second K-run
        float4 s0a = {0,0,0,0}, s0b = {0,0,0,0};
        float4 s1a = {0,0,0,0}, s1b = {0,0,0,0};
#pragma unroll
        for (int t = 0; t < 4; ++t) {
            const float* crow = Ch + (size_t)tk[t] * E_;   // PAD row 0 is all-zero in C
            acc4(s0a, ld4(crow + e0));  acc4(s0b, ld4(crow + e0 + 4));
            acc4(s1a, ld4(crow + e1));  acc4(s1b, ld4(crow + e1 + 4));
        }
        if (inconv) {
            acc4(s0a, ld4(rrow + e0));  acc4(s0b, ld4(rrow + e0 + 4));
            acc4(s1a, ld4(rrow + e1));  acc4(s1b, ld4(rrow + e1 + 4));
        }
        float tmp[16];
        *reinterpret_cast<float4*>(tmp + 0)  = s0a;
        *reinterpret_cast<float4*>(tmp + 4)  = s0b;
        *reinterpret_cast<float4*>(tmp + 8)  = s1a;
        *reinterpret_cast<float4*>(tmp + 12) = s1b;

        v16h afrag;
#pragma unroll
        for (int i = 0; i < 16; ++i) afrag[i] = (_Float16)(tmp[i] * gpv);

        // B fragment: q at K in [8h,8h+8) U [16+8h,16+8h+8) per lane-half (all N columns equal)
        float qt[16];
        *reinterpret_cast<float4*>(qt + 0)  = ld4(qb + e0);
        *reinterpret_cast<float4*>(qt + 4)  = ld4(qb + e0 + 4);
        *reinterpret_cast<float4*>(qt + 8)  = ld4(qb + e1);
        *reinterpret_cast<float4*>(qt + 12) = ld4(qb + e1 + 4);
        v16h bfrag;
#pragma unroll
        for (int i = 0; i < 16; ++i) bfrag[i] = (_Float16)qt[i];

        acc = __builtin_amdgcn_wmma_f32_16x16x32_f16(
                  false, afrag, false, bfrag, (short)0, acc, false, false);
    }

    // D layout: VGPR i holds (M=i, lanes 0-15) and (M=8+i, lanes 16-31).
    const int base = b * M_ + blockIdx.x * 128 + wave * 16;
    if (lane == 0) {
#pragma unroll
        for (int i = 0; i < 8; ++i) {
            prob_origin[base + i] = acc[i];
            if (out_origin) out_origin[base + i] = acc[i];
        }
    } else if (lane == 16) {
#pragma unroll
        for (int i = 0; i < 8; ++i) {
            prob_origin[base + 8 + i] = acc[i];
            if (out_origin) out_origin[base + 8 + i] = acc[i];
        }
    }
}

// ---------------------------------------------------------------------------
// softmax over M=2048 per batch; fixed-order LDS tree reduction (deterministic)
// ---------------------------------------------------------------------------
__global__ __launch_bounds__(256)
void softmax_kernel(const float* __restrict__ in,   // [B,M]
                    float*       __restrict__ out,  // [B,M] (ws)
                    float*       __restrict__ out2) // d_out half, or null
{
    const int b = blockIdx.x, tid = threadIdx.x;
    __shared__ float red[256];
    float v[8];
    float mx = -3.402823466e38f;
#pragma unroll
    for (int i = 0; i < 8; ++i) { v[i] = in[b * M_ + tid + i * 256]; mx = fmaxf(mx, v[i]); }
    red[tid] = mx; __syncthreads();
    for (int s = 128; s > 0; s >>= 1) {
        if (tid < s) red[tid] = fmaxf(red[tid], red[tid + s]);
        __syncthreads();
    }
    mx = red[0]; __syncthreads();
    float sm = 0.f;
#pragma unroll
    for (int i = 0; i < 8; ++i) { v[i] = __expf(v[i] - mx); sm += v[i]; }
    red[tid] = sm; __syncthreads();
    for (int s = 128; s > 0; s >>= 1) {
        if (tid < s) red[tid] += red[tid + s];
        __syncthreads();
    }
    const float inv = 1.f / red[0];
#pragma unroll
    for (int i = 0; i < 8; ++i) {
        const float p = v[i] * inv;
        out[b * M_ + tid + i * 256] = p;
        if (out2) out2[b * M_ + tid + i * 256] = p;
    }
}

// ---------------------------------------------------------------------------
// wsum_partial: partial o_k over a 128-row tile.
//   o_k[b,e] = sum_m prob[b,m]*gp[b,m]*(embsum_{h+1}[b,m,e]+conv)
// Each lane covers e in [4L,4L+4) and [128+4L,128+4L+4) (coalesced float4).
// Cross-wave combine via fixed-order LDS reduction -> partials[b, tile, e].
// ---------------------------------------------------------------------------
__global__ __launch_bounds__(256)
void wsum_partial_kernel(const float* __restrict__ Cn,     // C + (h+1)*V*E
                         const int*   __restrict__ story,
                         const int*   __restrict__ kb_len,
                         const int*   __restrict__ conv_len,
                         const float* __restrict__ rnn_out,
                         const float* __restrict__ gp,
                         const float* __restrict__ prob,   // softmax [B,M]
                         float*       __restrict__ partials) // [B,NT,E]
{
    const int b    = blockIdx.y;
    const int wave = threadIdx.x >> 5;
    const int lane = threadIdx.x & 31;
    const int m0   = blockIdx.x * 128 + wave * 16;
    const int ea   = lane * 4;
    const int eb   = 128 + lane * 4;
    const int kbl  = kb_len[b];
    const int cl   = conv_len[b];

    float4 s0 = {0,0,0,0}, s1 = {0,0,0,0};
    for (int r = 0; r < 16; ++r) {
        const int m = m0 + r;
        const float w = prob[b * M_ + m] * gp[b * M_ + m];
        const int4 tv = *reinterpret_cast<const int4*>(story + ((size_t)b * M_ + m) * T_);
        const int tk[4] = { tv.x, tv.y, tv.z, tv.w };
        float4 a0 = {0,0,0,0}, a1 = {0,0,0,0};
#pragma unroll
        for (int t = 0; t < 4; ++t) {
            const float* crow = Cn + (size_t)tk[t] * E_;
            acc4(a0, ld4(crow + ea));
            acc4(a1, ld4(crow + eb));
        }
        const int cidx = m - kbl;
        if (cidx >= 0 && cidx < cl) {
            const float* rrow = rnn_out + ((size_t)b * CM_ + cidx) * E_;
            acc4(a0, ld4(rrow + ea));
            acc4(a1, ld4(rrow + eb));
        }
        s0.x = fmaf(w, a0.x, s0.x); s0.y = fmaf(w, a0.y, s0.y);
        s0.z = fmaf(w, a0.z, s0.z); s0.w = fmaf(w, a0.w, s0.w);
        s1.x = fmaf(w, a1.x, s1.x); s1.y = fmaf(w, a1.y, s1.y);
        s1.z = fmaf(w, a1.z, s1.z); s1.w = fmaf(w, a1.w, s1.w);
    }

    __shared__ float red[8][E_];
    *reinterpret_cast<float4*>(&red[wave][ea]) = s0;
    *reinterpret_cast<float4*>(&red[wave][eb]) = s1;
    __syncthreads();
    const int e = threadIdx.x;          // 0..255
    float sum = 0.f;
#pragma unroll
    for (int w = 0; w < 8; ++w) sum += red[w][e];
    partials[((size_t)b * NT_ + blockIdx.x) * E_ + e] = sum;
}

// q_next[b,e] = q_cur[b,e] + sum_tile partials[b,tile,e]   (deterministic)
__global__ __launch_bounds__(256)
void wsum_reduce_kernel(const float* __restrict__ partials,
                        const float* __restrict__ q_cur,
                        float*       __restrict__ q_next)
{
    const int b = blockIdx.x, e = threadIdx.x;
    float s = q_cur[b * E_ + e];
#pragma unroll
    for (int t = 0; t < NT_; ++t) s += partials[((size_t)b * NT_ + t) * E_ + e];
    q_next[b * E_ + e] = s;
}

// ---------------------------------------------------------------------------
extern "C" void kernel_launch(void* const* d_in, const int* in_sizes, int n_in,
                              void* d_out, int out_size, void* d_ws, size_t ws_size,
                              hipStream_t stream)
{
    const int*   story      = (const int*)  d_in[0];
    const int*   kb_len     = (const int*)  d_in[1];
    const int*   conv_len   = (const int*)  d_in[2];
    const float* rnn_output = (const float*)d_in[3];
    const float* rnn_hidden = (const float*)d_in[4];
    const float* gp         = (const float*)d_in[5];
    const float* C          = (const float*)d_in[6];
    float*       out        = (float*)d_out;          // [prob_origin_f | prob_soft]

    // Workspace layout (floats): ~1.09 MB total
    float* ws          = (float*)d_ws;
    float* prob_origin = ws;                          // B*M
    float* prob_soft   = prob_origin + B_ * M_;       // B*M
    float* q1          = prob_soft   + B_ * M_;       // B*E
    float* q2          = q1          + B_ * E_;       // B*E
    float* partials    = q2          + B_ * E_;       // B*NT*E

    const size_t CH = (size_t)V_ * E_;
    const dim3 gTile(NT_, B_), gB(B_), blk(256);

    // ---- hop 0 (q = rnn_hidden) ----
    dot_kernel<<<gTile, blk, 0, stream>>>(C + 0 * CH, story, kb_len, conv_len,
                                          rnn_output, gp, rnn_hidden, prob_origin, nullptr);
    softmax_kernel<<<gB, blk, 0, stream>>>(prob_origin, prob_soft, nullptr);
    wsum_partial_kernel<<<gTile, blk, 0, stream>>>(C + 1 * CH, story, kb_len, conv_len,
                                                   rnn_output, gp, prob_soft, partials);
    wsum_reduce_kernel<<<gB, blk, 0, stream>>>(partials, rnn_hidden, q1);

    // ---- hop 1 ----
    dot_kernel<<<gTile, blk, 0, stream>>>(C + 1 * CH, story, kb_len, conv_len,
                                          rnn_output, gp, q1, prob_origin, nullptr);
    softmax_kernel<<<gB, blk, 0, stream>>>(prob_origin, prob_soft, nullptr);
    wsum_partial_kernel<<<gTile, blk, 0, stream>>>(C + 2 * CH, story, kb_len, conv_len,
                                                   rnn_output, gp, prob_soft, partials);
    wsum_reduce_kernel<<<gB, blk, 0, stream>>>(partials, q1, q2);

    // ---- hop 2 (outputs) ----
    dot_kernel<<<gTile, blk, 0, stream>>>(C + 2 * CH, story, kb_len, conv_len,
                                          rnn_output, gp, q2, prob_origin, out);
    softmax_kernel<<<gB, blk, 0, stream>>>(prob_origin, prob_soft, out + B_ * M_);
}
